// CausalSelfAttention_70471823393149
// MI455X (gfx1250) — compile-verified, matrix operand order checked by reading, not run on hardware
//
#include <hip/hip_runtime.h>
#include <hip/hip_bf16.h>

#define Bsz 8
#define Tsz 1024
#define Csz 768
#define Hsz 12
#define Dsz 64
#define BT  (Bsz * Tsz)       // 8192
#define C3  (3 * Csz)         // 2304
#define BH  (Bsz * Hsz)       // 96

typedef __attribute__((ext_vector_type(16))) __bf16 v16bf;
typedef __attribute__((ext_vector_type(8)))  float  v8f;

union Frag { v16bf v; uint4 q[2]; };

static __device__ __forceinline__ unsigned short f2bf(float f) {
    unsigned u = __float_as_uint(f);
    unsigned r = (u + 0x7FFFu + ((u >> 16) & 1u)) >> 16;
    return (unsigned short)r;
}

// ---------------------------------------------------------------------------
// fp32 -> bf16 (round to nearest even)
// ---------------------------------------------------------------------------
__global__ void cvt_bf16_kernel(const float* __restrict__ in,
                                unsigned short* __restrict__ out, int n) {
    int i = blockIdx.x * blockDim.x + threadIdx.x;
    if (i < n) out[i] = f2bf(in[i]);
}

// ---------------------------------------------------------------------------
// bf16 GEMM: C[M,N] = A[M,K] @ Bw[N,K]^T, one wave computes a 32x64 tile
// (2 M-subtiles x 4 N-subtiles -> 8 WMMAs per 12 b128 loads per K-step).
// mode 0: fused QKV epilogue (RoPE on q/k, q scaled 1/sqrt(D), v transposed)
// mode 1: plain f32 store to c_out
// ---------------------------------------------------------------------------
__global__ __launch_bounds__(32)
void gemm_bf16_kernel(const unsigned short* __restrict__ A,
                      const unsigned short* __restrict__ Bw,
                      int M, int N, int K,
                      unsigned short* __restrict__ q_out,
                      unsigned short* __restrict__ k_out,
                      unsigned short* __restrict__ vt_out,
                      float* __restrict__ c_out, int mode) {
    const int lane = threadIdx.x & 31;
    const int l16  = lane & 15;
    const int bh   = (lane >> 4) << 3;   // K-half offset in A/B layout: 0 or 8
    const int rh   = bh;                 // row-half offset in C layout: 0 or 8
    const int m0   = blockIdx.y * 32;
    const int n0   = blockIdx.x * 64;

    v8f acc[2][4] = {};

    const unsigned short* arow[2] = {
        A + (size_t)(m0 + l16) * K,
        A + (size_t)(m0 + 16 + l16) * K };
    const unsigned short* brows[4] = {
        Bw + (size_t)(n0 + 0 * 16 + l16) * K,
        Bw + (size_t)(n0 + 1 * 16 + l16) * K,
        Bw + (size_t)(n0 + 2 * 16 + l16) * K,
        Bw + (size_t)(n0 + 3 * 16 + l16) * K };

    for (int k = 0; k < K; k += 32) {
        Frag a[2];
#pragma unroll
        for (int mi = 0; mi < 2; ++mi) {
            a[mi].q[0] = *(const uint4*)(arow[mi] + k + bh);
            a[mi].q[1] = *(const uint4*)(arow[mi] + k + bh + 16);
            __builtin_prefetch(arow[mi] + k + 64, 0, 1);
        }
#pragma unroll
        for (int nt = 0; nt < 4; ++nt) {
            Frag b;
            b.q[0] = *(const uint4*)(brows[nt] + k + bh);
            b.q[1] = *(const uint4*)(brows[nt] + k + bh + 16);
#pragma unroll
            for (int mi = 0; mi < 2; ++mi)
                acc[mi][nt] = __builtin_amdgcn_wmma_f32_16x16x32_bf16(
                    false, a[mi].v, false, b.v, (short)0, acc[mi][nt],
                    false, false);
        }
    }

    if (mode == 0) {
        const int sect = n0 / Csz;          // 0=q, 1=k, 2=v
        const int h    = (n0 - sect * Csz) / Dsz;
        if (sect < 2) {
            unsigned short* dst = (sect == 0) ? q_out : k_out;
#pragma unroll
            for (int mi = 0; mi < 2; ++mi) {
#pragma unroll
                for (int j = 0; j < 2; ++j) {
                    const int i = j * 16 + l16;      // freq index = d (< 32)
                    const float invf = __powf(10000.f, -(float)(2 * i) / 64.f);
#pragma unroll
                    for (int r = 0; r < 8; ++r) {
                        const int m = m0 + mi * 16 + r + rh;
                        const int t = m & (Tsz - 1);
                        const int b = m >> 10;
                        float sn, cs;
                        __sincosf((float)t * invf, &sn, &cs);
                        float lo = acc[mi][j][r], hi = acc[mi][j + 2][r];
                        float nlo =  lo * cs + hi * sn;
                        float nhi = -lo * sn + hi * cs;
                        if (sect == 0) { nlo *= 0.125f; nhi *= 0.125f; }
                        size_t base = ((size_t)(b * Hsz + h) * Tsz + t) * Dsz;
                        dst[base + i]      = f2bf(nlo);
                        dst[base + 32 + i] = f2bf(nhi);
                    }
                }
            }
        } else {
#pragma unroll
            for (int mi = 0; mi < 2; ++mi) {
#pragma unroll
                for (int nt = 0; nt < 4; ++nt) {
                    const int d = nt * 16 + l16;
#pragma unroll
                    for (int r = 0; r < 8; ++r) {
                        const int m = m0 + mi * 16 + r + rh;
                        const int t = m & (Tsz - 1);
                        const int b = m >> 10;
                        vt_out[((size_t)(b * Hsz + h) * Dsz + d) * Tsz + t] =
                            f2bf(acc[mi][nt][r]);
                    }
                }
            }
        }
    } else {
#pragma unroll
        for (int mi = 0; mi < 2; ++mi) {
#pragma unroll
            for (int nt = 0; nt < 4; ++nt) {
                const int n = n0 + nt * 16 + l16;
#pragma unroll
                for (int r = 0; r < 8; ++r) {
                    const int m = m0 + mi * 16 + r + rh;
                    c_out[(size_t)m * N + n] = acc[mi][nt][r];
                }
            }
        }
    }
}

// ---------------------------------------------------------------------------
// Flash attention: 1 wave per (bh, 32-query tile); key blocks of 32.
// Two 16-query subtiles share every K and V fragment (halves K/V traffic).
// Q,K: bf16 [BH, T, D] (q pre-scaled, both RoPE'd). Vt: bf16 [BH, D, T].
// Y: bf16 [B, T, C].
// ---------------------------------------------------------------------------
__global__ __launch_bounds__(32)
void attn_kernel(const unsigned short* __restrict__ Q,
                 const unsigned short* __restrict__ Km,
                 const unsigned short* __restrict__ Vt,
                 unsigned short* __restrict__ Y) {
    __shared__ unsigned short plds[32 * 32];

    const int lane = threadIdx.x & 31;
    const int l16  = lane & 15;
    const int bh   = (lane >> 4) << 3;   // 0 or 8 (layout K-half / row-half)
    const int m0   = blockIdx.x * 32;
    const int bhid = blockIdx.y;

    // Q fragments: 2 query subtiles x 2 K-dim steps (d = 0..63)
    Frag qa[2][2];
#pragma unroll
    for (int qt = 0; qt < 2; ++qt) {
        const unsigned short* qrow =
            Q + ((size_t)bhid * Tsz + (m0 + qt * 16 + l16)) * Dsz;
#pragma unroll
        for (int kk = 0; kk < 2; ++kk) {
            qa[qt][kk].q[0] = *(const uint4*)(qrow + kk * 32 + bh);
            qa[qt][kk].q[1] = *(const uint4*)(qrow + kk * 32 + bh + 16);
        }
    }

    v8f o[2][4] = {};
    float mrow[2][8], lrow[2][8];
#pragma unroll
    for (int qt = 0; qt < 2; ++qt)
#pragma unroll
        for (int r = 0; r < 8; ++r) { mrow[qt][r] = -3.0e38f; lrow[qt][r] = 0.f; }

    const int kb_end = m0 + 32;          // need keys <= m0+31
    for (int kb = 0; kb < kb_end; kb += 32) {
        // S = Q[32x64] @ K^T[64x32]; K fragments shared by both query subtiles
        v8f s[2][2] = {};
        const unsigned short* k0row =
            Km + ((size_t)bhid * Tsz + (kb + l16)) * Dsz;
        const unsigned short* k1row = k0row + 16 * Dsz;
#pragma unroll
        for (int kk = 0; kk < 2; ++kk) {
            Frag b0, b1;
            b0.q[0] = *(const uint4*)(k0row + kk * 32 + bh);
            b0.q[1] = *(const uint4*)(k0row + kk * 32 + bh + 16);
            b1.q[0] = *(const uint4*)(k1row + kk * 32 + bh);
            b1.q[1] = *(const uint4*)(k1row + kk * 32 + bh + 16);
#pragma unroll
            for (int qt = 0; qt < 2; ++qt) {
                s[qt][0] = __builtin_amdgcn_wmma_f32_16x16x32_bf16(
                    false, qa[qt][kk].v, false, b0.v, (short)0, s[qt][0],
                    false, false);
                s[qt][1] = __builtin_amdgcn_wmma_f32_16x16x32_bf16(
                    false, qa[qt][kk].v, false, b1.v, (short)0, s[qt][1],
                    false, false);
            }
        }

        // causal mask + online softmax (row lives across a 16-lane group)
        const int col0 = kb + l16, col1 = col0 + 16;
#pragma unroll
        for (int qt = 0; qt < 2; ++qt) {
#pragma unroll
            for (int r = 0; r < 8; ++r) {
                const int row = m0 + qt * 16 + r + bh;
                float v0 = (col0 <= row) ? s[qt][0][r] : -3.0e38f;
                float v1 = (col1 <= row) ? s[qt][1][r] : -3.0e38f;
                float mx = fmaxf(v0, v1);
#pragma unroll
                for (int off = 8; off >= 1; off >>= 1)
                    mx = fmaxf(mx, __shfl_xor(mx, off, 32));
                const float mn = fmaxf(mrow[qt][r], mx);
                const float alpha = __expf(mrow[qt][r] - mn);
                mrow[qt][r] = mn;
                const float p0 = __expf(v0 - mn);
                const float p1 = __expf(v1 - mn);
                float rs = p0 + p1;
#pragma unroll
                for (int off = 8; off >= 1; off >>= 1)
                    rs += __shfl_xor(rs, off, 32);
                lrow[qt][r] = lrow[qt][r] * alpha + rs;
#pragma unroll
                for (int nt = 0; nt < 4; ++nt) o[qt][nt][r] *= alpha;
                // stash P (bf16) row-major in LDS for C-layout -> A-layout
                plds[(qt * 16 + r + bh) * 32 + l16]      = f2bf(p0);
                plds[(qt * 16 + r + bh) * 32 + 16 + l16] = f2bf(p1);
            }
        }
        asm volatile("s_wait_dscnt 0" ::: "memory");

        // reload P in A-matrix layout (two 16x32 operands, K = key)
        Frag pa[2];
#pragma unroll
        for (int qt = 0; qt < 2; ++qt) {
            pa[qt].q[0] = *(const uint4*)(&plds[(qt * 16 + l16) * 32 + bh]);
            pa[qt].q[1] = *(const uint4*)(&plds[(qt * 16 + l16) * 32 + bh + 16]);
        }

        // O += P[32x32] @ V[32x64]; V fragments shared by both query subtiles
#pragma unroll
        for (int nt = 0; nt < 4; ++nt) {
            const unsigned short* vrow =
                Vt + ((size_t)bhid * Dsz + nt * 16 + l16) * Tsz + kb;
            Frag vb;
            vb.q[0] = *(const uint4*)(vrow + bh);
            vb.q[1] = *(const uint4*)(vrow + bh + 16);
#pragma unroll
            for (int qt = 0; qt < 2; ++qt)
                o[qt][nt] = __builtin_amdgcn_wmma_f32_16x16x32_bf16(
                    false, pa[qt].v, false, vb.v, (short)0, o[qt][nt],
                    false, false);
        }
    }

    // normalize & store Y (bf16 [B,T,C])
    const int h = bhid % Hsz, b = bhid / Hsz;
#pragma unroll
    for (int qt = 0; qt < 2; ++qt) {
#pragma unroll
        for (int r = 0; r < 8; ++r) {
            const int t = m0 + qt * 16 + r + bh;
            const float inv = 1.f / lrow[qt][r];
            size_t base = ((size_t)(b * Tsz + t)) * Csz + h * Dsz;
#pragma unroll
            for (int nt = 0; nt < 4; ++nt)
                Y[base + nt * 16 + l16] = f2bf(o[qt][nt][r] * inv);
        }
    }
}

// ---------------------------------------------------------------------------
extern "C" void kernel_launch(void* const* d_in, const int* in_sizes, int n_in,
                              void* d_out, int out_size, void* d_ws, size_t ws_size,
                              hipStream_t stream) {
    const float* x      = (const float*)d_in[0];   // [B,T,C]
    const float* w_qkv  = (const float*)d_in[1];   // [3C,C]
    const float* w_proj = (const float*)d_in[2];   // [C,C]
    float* out = (float*)d_out;                    // [B,T,C]

    char* ws = (char*)d_ws;
    size_t off = 0;
    auto take = [&](size_t bytes) -> void* {
        void* p = ws + off;
        off += (bytes + 255) & ~(size_t)255;
        return p;
    };
    unsigned short* xb  = (unsigned short*)take((size_t)BT * Csz * 2);
    unsigned short* wqb = (unsigned short*)take((size_t)C3 * Csz * 2);
    unsigned short* wpb = (unsigned short*)take((size_t)Csz * Csz * 2);
    unsigned short* qb  = (unsigned short*)take((size_t)BH * Tsz * Dsz * 2);
    unsigned short* kb  = (unsigned short*)take((size_t)BH * Tsz * Dsz * 2);
    unsigned short* vtb = (unsigned short*)take((size_t)BH * Dsz * Tsz * 2);
    unsigned short* yb  = (unsigned short*)take((size_t)BT * Csz * 2);

    // 1) fp32 -> bf16
    {
        int n = BT * Csz;
        cvt_bf16_kernel<<<(n + 255) / 256, 256, 0, stream>>>(x, xb, n);
        n = C3 * Csz;
        cvt_bf16_kernel<<<(n + 255) / 256, 256, 0, stream>>>(w_qkv, wqb, n);
        n = Csz * Csz;
        cvt_bf16_kernel<<<(n + 255) / 256, 256, 0, stream>>>(w_proj, wpb, n);
    }

    // 2) fused QKV projection + RoPE + V transpose
    {
        dim3 grid(C3 / 64, BT / 32);   // (36, 256)
        gemm_bf16_kernel<<<grid, 32, 0, stream>>>(
            xb, wqb, BT, C3, Csz, qb, kb, vtb, nullptr, 0);
    }

    // 3) flash attention (32 queries per wave)
    {
        dim3 grid(Tsz / 32, BH);       // (32, 96)
        attn_kernel<<<grid, 32, 0, stream>>>(qb, kb, vtb, yb);
    }

    // 4) output projection
    {
        dim3 grid(Csz / 64, BT / 32);  // (12, 256)
        gemm_bf16_kernel<<<grid, 32, 0, stream>>>(
            yb, wpb, BT, Csz, Csz, nullptr, nullptr, nullptr, out, 1);
    }
}
